// MultiheadAttention_42460046688787
// MI455X (gfx1250) — matrix operand order repaired and checked by reading, not compiled, not run on hardware
//
#include <hip/hip_runtime.h>
#include <hip/hip_bf16.h>

// Problem constants (reference: T=2048, B=4, E=768, H=12, Dh=64)
#define TT 2048
#define BZ 4
#define EE 768
#define HH 12
#define DH 64
#define MM (TT * BZ)   // 8192 rows
#define N1 (3 * EE)    // 2304 qkv cols

typedef __attribute__((ext_vector_type(16))) __bf16 v16bf;
typedef __attribute__((ext_vector_type(8)))  __bf16 v8bf;
typedef __attribute__((ext_vector_type(8)))  float  v8f;
typedef unsigned short u16;

union ABf { v16bf v; v8bf h[2]; };

__device__ __forceinline__ u16 f2bfbits(float f) {
  union { float f; unsigned u; } c; c.f = f;
  return (u16)((c.u + 0x7FFFu + ((c.u >> 16) & 1u)) >> 16);  // RNE
}

__device__ __forceinline__ v8bf ldv8(const u16* p) {
  return *reinterpret_cast<const v8bf*>(p);
}

__device__ __forceinline__ v8f wmma_bf16(const v16bf& a, const v16bf& b, const v8f& c) {
  // (neg_a, A, neg_b, B, c_mod, C, reuse_a, reuse_b)
  return __builtin_amdgcn_wmma_f32_16x16x32_bf16(false, a, false, b, (short)0, c,
                                                 false, false);
}

// ---------------------------------------------------------------- f32 -> bf16
__global__ void __launch_bounds__(256) cvt_bf16(const float* __restrict__ s,
                                                u16* __restrict__ d, int n) {
  for (int i = blockIdx.x * blockDim.x + threadIdx.x; i < n;
       i += gridDim.x * blockDim.x)
    d[i] = f2bfbits(s[i]);
}

// -------------------------------------------------- QKV GEMM (8192x2304x768)
// qkv[m,n] = sum_k x[m,k] * Win[n,k] + bias[n]; scatter into Q/K/V^T bf16.
__global__ void __launch_bounds__(256) qkv_gemm(
    const u16* __restrict__ xb, const u16* __restrict__ wb,
    const float* __restrict__ bias,
    u16* __restrict__ q, u16* __restrict__ k, u16* __restrict__ vt) {
  const int lane = threadIdx.x & 31;
  const int w    = threadIdx.x >> 5;
  const int ln = lane & 15, hf = lane >> 4;
  const int m0 = blockIdx.y * 128 + (w & 3) * 32;   // 4 waves along M
  const int n0 = blockIdx.x * 128 + (w >> 2) * 64;  // 2 waves along N
  v8f c[2][4] = {};
  for (int k0 = 0; k0 < EE; k0 += 32) {
    ABf a[2], b[4];
#pragma unroll
    for (int mm = 0; mm < 2; ++mm) {
      const u16* ap = xb + (size_t)(m0 + mm * 16 + ln) * EE + k0;
      a[mm].h[0] = ldv8(ap + 8 * hf);        // K = k0+8h .. +7
      a[mm].h[1] = ldv8(ap + 16 + 8 * hf);   // K = k0+16+8h .. +7
    }
#pragma unroll
    for (int nn = 0; nn < 4; ++nn) {
      const u16* bp = wb + (size_t)(n0 + nn * 16 + ln) * EE + k0;
      b[nn].h[0] = ldv8(bp + 8 * hf);        // K = k0+8h .. +7
      b[nn].h[1] = ldv8(bp + 16 + 8 * hf);   // K = k0+16+8h .. +7
    }
#pragma unroll
    for (int mm = 0; mm < 2; ++mm)
#pragma unroll
      for (int nn = 0; nn < 4; ++nn)
        c[mm][nn] = wmma_bf16(a[mm].v, b[nn].v, c[mm][nn]);
  }
  // Epilogue: bias, Q scaling, head-layout scatter (bf16)
#pragma unroll
  for (int nn = 0; nn < 4; ++nn) {
    const int n = n0 + nn * 16 + ln;
    const float bn = bias[n];
    const int sec = n / EE;           // 0=Q 1=K 2=V (uniform per 128-col block)
    const int ne = n - sec * EE;
    const int hh = ne >> 6, d = ne & 63;
#pragma unroll
    for (int mm = 0; mm < 2; ++mm)
#pragma unroll
      for (int i = 0; i < 8; ++i) {
        const int m = m0 + mm * 16 + i + 8 * hf;  // C layout: row = i + 8*half
        const int t = m >> 2, bz = m & 3;         // row m = t*B + b
        const float val = c[mm][nn][i] + bn;
        if (sec == 0)
          q[((size_t)(bz * HH + hh) * TT + t) * DH + d] = f2bfbits(val * 0.125f);
        else if (sec == 1)
          k[((size_t)(bz * HH + hh) * TT + t) * DH + d] = f2bfbits(val);
        else
          vt[((size_t)(bz * HH + hh) * DH + d) * TT + t] = f2bfbits(val);
      }
  }
}

// ----------------------------------------- flash attention, 16 rows per wave
__global__ void __launch_bounds__(64) attn_flash(
    const u16* __restrict__ q, const u16* __restrict__ kk,
    const u16* __restrict__ vt, const float* __restrict__ abias,
    const unsigned char* __restrict__ kpm, u16* __restrict__ aws) {
  __shared__ __align__(16) u16 lds[2][16 * 32];  // per-wave P staging tile
  const int lane = threadIdx.x & 31;
  const int w    = threadIdx.x >> 5;
  const int ln = lane & 15, hf = lane >> 4;
  const int t0 = (blockIdx.x * 2 + w) * 16;
  const int hh = blockIdx.y, bz = blockIdx.z;
  const u16* qb = q  + (size_t)(bz * HH + hh) * TT * DH;
  const u16* kb = kk + (size_t)(bz * HH + hh) * TT * DH;
  const u16* vb = vt + (size_t)(bz * HH + hh) * DH * TT;

  // Q A-fragments (16 rows x 64 d = two 16x32 frags), cached for whole loop
  ABf qa[2];
  {
    const u16* qp = qb + (size_t)(t0 + ln) * DH;
    qa[0].h[0] = ldv8(qp + 8 * hf);
    qa[0].h[1] = ldv8(qp + 16 + 8 * hf);
    qa[1].h[0] = ldv8(qp + 32 + 8 * hf);
    qa[1].h[1] = ldv8(qp + 48 + 8 * hf);
  }
  v8f o[4] = {};
  float mrow[8], lrow[8];
#pragma unroll
  for (int i = 0; i < 8; ++i) { mrow[i] = -1e30f; lrow[i] = 0.f; }

  for (int s0 = 0; s0 < TT; s0 += 32) {
    // ---- scores S = (Q*scaling) @ K^T for 16x32 block, + bias, + mask
    v8f s[2] = {};
#pragma unroll
    for (int half = 0; half < 2; ++half) {
      const int sc = s0 + half * 16 + ln;
      const u16* kp = kb + (size_t)sc * DH;
      ABf b0, b1;
      b0.h[0] = ldv8(kp + 8 * hf);             // d = 8h.., 16+8h..
      b0.h[1] = ldv8(kp + 16 + 8 * hf);
      b1.h[0] = ldv8(kp + 32 + 8 * hf);        // d = 32+8h.., 48+8h..
      b1.h[1] = ldv8(kp + 48 + 8 * hf);
      s[half] = wmma_bf16(qa[0].v, b0.v, s[half]);
      s[half] = wmma_bf16(qa[1].v, b1.v, s[half]);
      const float bv = abias[sc * BZ + bz];
      const bool  mk = kpm[sc * BZ + bz] != 0;
#pragma unroll
      for (int i = 0; i < 8; ++i)              // cndmask: EXEC untouched
        s[half][i] = mk ? -1e30f : (s[half][i] + bv);
    }
    // ---- online softmax: row stats via 16-lane xor-shuffle reductions
    float scl[8];
#pragma unroll
    for (int i = 0; i < 8; ++i) {
      float bm = fmaxf(s[0][i], s[1][i]);
#pragma unroll
      for (int x = 1; x < 16; x <<= 1) bm = fmaxf(bm, __shfl_xor(bm, x, 32));
      const float mn = fmaxf(mrow[i], bm);
      const float sc = __expf(mrow[i] - mn);
      mrow[i] = mn;
      const float p0 = __expf(s[0][i] - mn);
      const float p1 = __expf(s[1][i] - mn);
      s[0][i] = p0; s[1][i] = p1;
      float rs = p0 + p1;
#pragma unroll
      for (int x = 1; x < 16; x <<= 1) rs += __shfl_xor(rs, x, 32);
      lrow[i] = lrow[i] * sc + rs;
      scl[i] = sc;
    }
#pragma unroll
    for (int f = 0; f < 4; ++f)
#pragma unroll
      for (int i = 0; i < 8; ++i) o[f][i] *= scl[i];
    // ---- restage P through LDS: C-layout -> row-major [16][32] -> A-layout
    u16* pl = lds[w];
#pragma unroll
    for (int i = 0; i < 8; ++i) {
      const int r = i + 8 * hf;
      pl[r * 32 + ln]      = f2bfbits(s[0][i]);
      pl[r * 32 + 16 + ln] = f2bfbits(s[1][i]);
    }
    asm volatile("s_wait_dscnt 0x0" ::: "memory");  // CDNA5 split DS counter
    __syncthreads();
    ABf pa;
    {
      const u16* pr = pl + ln * 32;
      pa.h[0] = ldv8(pr + 8 * hf);
      pa.h[1] = ldv8(pr + 16 + 8 * hf);
    }
    // ---- O += P @ V  (V^T layout gives contiguous B-fragment loads)
#pragma unroll
    for (int f = 0; f < 4; ++f) {
      const u16* vp = vb + (size_t)(f * 16 + ln) * TT + s0;
      ABf vf;
      vf.h[0] = ldv8(vp + 8 * hf);
      vf.h[1] = ldv8(vp + 16 + 8 * hf);
      o[f] = wmma_bf16(pa.v, vf.v, o[f]);
    }
    __syncthreads();
  }
  // ---- normalize and emit attn rows as bf16 GEMM operand [m][e]
#pragma unroll
  for (int i = 0; i < 8; ++i) {
    const float inv = lrow[i] > 0.f ? 1.f / lrow[i] : 0.f;
    const int t = t0 + i + 8 * hf;
#pragma unroll
    for (int f = 0; f < 4; ++f) {
      const int d = f * 16 + ln;
      aws[((size_t)t * BZ + bz) * EE + hh * DH + d] = f2bfbits(o[f][i] * inv);
    }
  }
}

// -------------------------------------------- out projection (8192x768x768)
__global__ void __launch_bounds__(256) out_gemm(
    const u16* __restrict__ ab, const u16* __restrict__ wb,
    const float* __restrict__ bias, float* __restrict__ out) {
  const int lane = threadIdx.x & 31;
  const int w    = threadIdx.x >> 5;
  const int ln = lane & 15, hf = lane >> 4;
  const int m0 = blockIdx.y * 128 + (w & 3) * 32;
  const int n0 = blockIdx.x * 128 + (w >> 2) * 64;
  v8f c[2][4] = {};
  for (int k0 = 0; k0 < EE; k0 += 32) {
    ABf a[2], b[4];
#pragma unroll
    for (int mm = 0; mm < 2; ++mm) {
      const u16* ap = ab + (size_t)(m0 + mm * 16 + ln) * EE + k0;
      a[mm].h[0] = ldv8(ap + 8 * hf);
      a[mm].h[1] = ldv8(ap + 16 + 8 * hf);
    }
#pragma unroll
    for (int nn = 0; nn < 4; ++nn) {
      const u16* bp = wb + (size_t)(n0 + nn * 16 + ln) * EE + k0;
      b[nn].h[0] = ldv8(bp + 8 * hf);
      b[nn].h[1] = ldv8(bp + 16 + 8 * hf);
    }
#pragma unroll
    for (int mm = 0; mm < 2; ++mm)
#pragma unroll
      for (int nn = 0; nn < 4; ++nn)
        c[mm][nn] = wmma_bf16(a[mm].v, b[nn].v, c[mm][nn]);
  }
#pragma unroll
  for (int nn = 0; nn < 4; ++nn) {
    const int n = n0 + nn * 16 + ln;
    const float bn = bias[n];
#pragma unroll
    for (int mm = 0; mm < 2; ++mm)
#pragma unroll
      for (int i = 0; i < 8; ++i) {
        const int m = m0 + mm * 16 + i + 8 * hf;
        out[(size_t)m * EE + n] = c[mm][nn][i] + bn;
      }
  }
}

extern "C" void kernel_launch(void* const* d_in, const int* in_sizes, int n_in,
                              void* d_out, int out_size, void* d_ws, size_t ws_size,
                              hipStream_t stream) {
  (void)in_sizes; (void)n_in; (void)out_size; (void)ws_size;
  const float*         x     = (const float*)d_in[0];
  const unsigned char* kpm   = (const unsigned char*)d_in[1];  // bool mask (T,B)
  const float*         abias = (const float*)d_in[2];          // (T,B)
  const float*         win   = (const float*)d_in[3];          // (2304,768)
  const float*         binb  = (const float*)d_in[4];          // (2304,)
  const float*         wout  = (const float*)d_in[5];          // (768,768)
  const float*         bout  = (const float*)d_in[6];          // (768,)

  // Workspace carve-up (bf16 elements)
  u16* p = (u16*)d_ws;
  u16* xb    = p; p += (size_t)MM * EE;             // x as bf16 rows
  u16* winb  = p; p += (size_t)N1 * EE;             // in_proj_weight bf16
  u16* woutb = p; p += (size_t)EE * EE;             // out_weight bf16
  u16* qws   = p; p += (size_t)BZ * HH * TT * DH;   // Q [b][h][t][d] (scaled)
  u16* kws   = p; p += (size_t)BZ * HH * TT * DH;   // K [b][h][s][d]
  u16* vtws  = p; p += (size_t)BZ * HH * DH * TT;   // V^T [b][h][d][s]
  u16* aws   = p; p += (size_t)MM * EE;             // attn output bf16 rows

  cvt_bf16<<<4096, 256, 0, stream>>>(x,    xb,    MM * EE);
  cvt_bf16<<<2048, 256, 0, stream>>>(win,  winb,  N1 * EE);
  cvt_bf16<<<1024, 256, 0, stream>>>(wout, woutb, EE * EE);

  qkv_gemm<<<dim3(N1 / 128, MM / 128), 256, 0, stream>>>(xb, winb, binb,
                                                         qws, kws, vtws);
  attn_flash<<<dim3(TT / 32, HH, BZ), 64, 0, stream>>>(qws, kws, vtws,
                                                       abias, kpm, aws);
  out_gemm<<<dim3(EE / 128, MM / 128), 256, 0, stream>>>(aws, woutb, bout,
                                                         (float*)d_out);
}
